// Agent_Memory_34780645163637
// MI455X (gfx1250) — matrix operand order repaired and checked by reading, not compiled, not run on hardware
//
#include <hip/hip_runtime.h>
#include <hip/hip_bf16.h>

typedef __bf16 bf16;
typedef __attribute__((ext_vector_type(16))) __bf16 v16bf;
typedef __attribute__((ext_vector_type(8)))  __bf16 v8bf;
typedef __attribute__((ext_vector_type(8)))  float  v8f;

union FragBF { v16bf v; struct { v8bf lo; v8bf hi; } h; };

#if defined(__has_builtin)
#  if __has_builtin(__builtin_amdgcn_global_load_async_to_lds_b128)
#    define ASYNC_LDS 1
#  endif
#  if __has_builtin(__builtin_amdgcn_tensor_load_to_lds) && \
      __has_builtin(__builtin_amdgcn_s_wait_tensorcnt)
#    define HAVE_TDM 1
#  endif
#endif
#ifndef ASYNC_LDS
#  define ASYNC_LDS 0
#endif
#ifndef HAVE_TDM
#  define HAVE_TDM 0
#endif

// exact parameter type reported by clang: 'int __attribute__((vector_size(16))) *'
typedef int v4i_ __attribute__((vector_size(16)));
typedef unsigned int u32x4 __attribute__((ext_vector_type(4)));
typedef int i32x8 __attribute__((ext_vector_type(8)));
typedef int i32x4 __attribute__((ext_vector_type(4)));

// move 16B global -> LDS (async-to-LDS when available, else through VGPRs)
__device__ __forceinline__ void stage16(const bf16* g, bf16* l) {
#if ASYNC_LDS
    __builtin_amdgcn_global_load_async_to_lds_b128(
        (v4i_*)g, (__attribute__((address_space(3))) v4i_*)l, 0, 0);
#else
    *(v8bf*)l = *(const v8bf*)g;
#endif
}
__device__ __forceinline__ void stage_zero16(bf16* l) {
    v8bf z = {};
    *(v8bf*)l = z;
}
__device__ __forceinline__ void stage_wait() {
#if ASYNC_LDS
#  if __has_builtin(__builtin_amdgcn_s_wait_asynccnt)
    __builtin_amdgcn_s_wait_asynccnt(0);
#  else
    asm volatile("s_wait_asynccnt 0x0" ::: "memory");
#  endif
#endif
}

#if HAVE_TDM
// LDS byte offset of a generic pointer into shared memory
__device__ __forceinline__ unsigned lds_off(const void* p) {
    return (unsigned)(size_t)(__attribute__((address_space(3))) const void*)p;
}
// TDM: DMA a 2D tile (tile0 x tile1 elements, 2B each, row stride stride0)
// from global 'gaddr' (tile start) into LDS at byte offset 'lds_addr'.
// dim0/dim1 are tensor extents relative to the tile start (OOB reads -> 0).
__device__ __forceinline__ void tdm_load_2d(unsigned lds_addr, const void* gaddr,
                                            unsigned dim0, unsigned dim1,
                                            unsigned stride0,
                                            unsigned tile0, unsigned tile1) {
    const unsigned long long ga = (unsigned long long)(size_t)gaddr;
    u32x4 g0 = { 1u,                                   // count=1, user descriptor
                 lds_addr,                             // bits 63:32  lds_addr
                 (unsigned)ga,                         // bits 95:64  global_addr lo
                 ((unsigned)(ga >> 32) & 0x01FFFFFFu)  // bits 120:96 global_addr hi
                   | 0x80000000u };                    // bits 127:126 type=2
    i32x8 g1 = { (int)(1u << 16),                           // data_size=1 (2B)
                 (int)((dim0 & 0xffffu) << 16),             // tensor_dim0[15:0]
                 (int)(((dim0 >> 16) & 0xffffu) |
                       ((dim1 & 0xffffu) << 16)),           // dim0 hi | dim1 lo
                 (int)(((dim1 >> 16) & 0xffffu) |
                       ((tile0 & 0xffffu) << 16)),          // dim1 hi | tile_dim0
                 (int)(tile1 & 0xffffu),                    // tile_dim1 (tile_dim2=0)
                 (int)stride0,                              // tensor_dim0_stride lo
                 0, 0 };
    i32x4 z4 = { 0, 0, 0, 0 };
    i32x8 z8 = { 0, 0, 0, 0, 0, 0, 0, 0 };
    __builtin_amdgcn_tensor_load_to_lds(g0, g1, z4, z4, z8, 0);
}
__device__ __forceinline__ void tdm_wait() {
    __builtin_amdgcn_s_wait_tensorcnt(0);
}
#endif

__device__ __forceinline__ float sigmoidf_(float x) { return 1.0f / (1.0f + expf(-x)); }

// ---------------------------------------------------------------------------
// f32 -> bf16 elementwise convert / vector add
// ---------------------------------------------------------------------------
__global__ void __launch_bounds__(256) cvt_f32_bf16_k(const float* __restrict__ in,
                                                      bf16* __restrict__ out, int n) {
    int i = blockIdx.x * 256 + threadIdx.x;
    if (i < n) out[i] = (bf16)in[i];
}

__global__ void __launch_bounds__(256) vec_add_k(const float* __restrict__ a,
                                                 const float* __restrict__ b,
                                                 float* __restrict__ o, int n) {
    int i = blockIdx.x * 256 + threadIdx.x;
    if (i < n) o[i] = a[i] + b[i];
}

// ---------------------------------------------------------------------------
// WMMA bf16 GEMM: C[M,N] = act(A[M,K] @ W[N,K]^T + bias[N])
// Block 256 thr (8 waves), tile 128x128, K chunks of 32, double-buffered LDS.
// A tile: per-lane async-to-LDS. W tile: one TDM descriptor per chunk
// (wave 0 issues; TENSORcnt waited before the block barrier).
// ---------------------------------------------------------------------------
#define BM 128
#define BN 128
#define BK 32

__global__ void __launch_bounds__(256)
gemm_bf16_wmma_k(const bf16* __restrict__ A, int lda,
                 const bf16* __restrict__ W, int ldw,
                 const float* __restrict__ bias,
                 float* __restrict__ outF,
                 bf16* __restrict__ outB,
                 int M, int N, int K, int actv)
{
    __shared__ __align__(16) bf16 sA[2][BM * BK];
    __shared__ __align__(16) bf16 sW[2][BN * BK];

    const int tid  = threadIdx.x;
    const int lane = tid & 31;
    const int wave = tid >> 5;
    const int bm = blockIdx.y * BM;
    const int bn = blockIdx.x * BN;
    const int wm = (wave & 3) * 32;
    const int wn = (wave >> 2) * 64;

    v8f zero = {};
    v8f acc[2][4];
#pragma unroll
    for (int t = 0; t < 2; ++t)
#pragma unroll
        for (int u = 0; u < 4; ++u) acc[t][u] = zero;

    auto stageA = [&](int buf, int k0) {
#pragma unroll
        for (int rep = 0; rep < 2; ++rep) {
            const int e   = tid + rep * 256;   // 0..511
            const int row = e >> 2;            // 0..127
            const int col = (e & 3) * 8;       // 0,8,16,24
            stage16(A + (size_t)(bm + row) * lda + k0 + col,
                    &sA[buf][row * BK + col]);
        }
    };
    auto stageW = [&](int buf, int k0) {
#if HAVE_TDM
        if (wave == 0)
            tdm_load_2d(lds_off(&sW[buf][0]),
                        W + (size_t)bn * ldw + k0,
                        (unsigned)(K - k0), (unsigned)(N - bn), (unsigned)ldw,
                        BK, BN);
#else
#pragma unroll
        for (int rep = 0; rep < 2; ++rep) {
            const int e   = tid + rep * 256;
            const int row = e >> 2;
            const int col = (e & 3) * 8;
            const int wrow = bn + row;
            if (wrow < N)
                stage16(W + (size_t)wrow * ldw + k0 + col,
                        &sW[buf][row * BK + col]);
            else
                stage_zero16(&sW[buf][row * BK + col]);
        }
#endif
    };
    auto xfer_wait = [&]() {
        stage_wait();
#if HAVE_TDM
        if (wave == 0) tdm_wait();
#endif
    };

    stageA(0, 0);
    stageW(0, 0);
    xfer_wait();
    __syncthreads();

    const int nk = K / BK;
    for (int i = 0; i < nk; ++i) {
        const int cur = i & 1;
        if (i + 1 < nk) { stageA(cur ^ 1, (i + 1) * BK); stageW(cur ^ 1, (i + 1) * BK); }

        // fragments per ISA 16-bit layouts
        const int ko = (lane >> 4) * 8;    // A: K base per lane half
        const int kh = (lane >> 4) * 8;    // B: K base per lane half
        FragBF a[2], b[4];
#pragma unroll
        for (int t = 0; t < 2; ++t) {
            const int m = wm + t * 16 + (lane & 15);
            a[t].h.lo = *(const v8bf*)&sA[cur][m * BK + ko];
            a[t].h.hi = *(const v8bf*)&sA[cur][m * BK + ko + 16];
        }
#pragma unroll
        for (int u = 0; u < 4; ++u) {
            const int n = wn + u * 16 + (lane & 15);
            b[u].h.lo = *(const v8bf*)&sW[cur][n * BK + kh];
            b[u].h.hi = *(const v8bf*)&sW[cur][n * BK + kh + 16];
        }
#pragma unroll
        for (int t = 0; t < 2; ++t)
#pragma unroll
            for (int u = 0; u < 4; ++u)
                acc[t][u] = __builtin_amdgcn_wmma_f32_16x16x32_bf16(
                    false, a[t].v, false, b[u].v, (short)0, acc[t][u], false, false);

        if (i + 1 < nk) xfer_wait();
        __syncthreads();
    }

    // epilogue: C layout VGPR r -> M = r + (lane>=16)*8, N = lane&15
    const int rbase = bm + wm + (lane >> 4) * 8;
#pragma unroll
    for (int t = 0; t < 2; ++t) {
#pragma unroll
        for (int u = 0; u < 4; ++u) {
            const int col = bn + wn + u * 16 + (lane & 15);
            if (col < N) {
                const float bia = bias ? bias[col] : 0.0f;
#pragma unroll
                for (int r = 0; r < 8; ++r) {
                    const int row = rbase + t * 16 + r;
                    float v = acc[t][u][r] + bia;
                    if (actv) v = tanhf(v);
                    const size_t o = (size_t)row * N + col;
                    if (outF) outF[o] = v;
                    if (outB) outB[o] = (bf16)v;
                }
            }
        }
    }
}

// ---------------------------------------------------------------------------
// Fused LSTM step (one t):
//   g = xg_t + (h_prev * (1-done)) @ w_hh^T   (WMMA, acc seeded from xg)
//   c = sig(f)*c*(1-done) + sig(i)*tanh(g_g);  h = sig(o)*tanh(c)
// Block 256 thr covers B=64 rows x 32 H-cols; grid.x = H/32. Double-buffered.
// W tile: 4 TDM descriptors (one per gate row-block) issued by wave 0.
// ---------------------------------------------------------------------------
__global__ void __launch_bounds__(256)
lstm_step_k(const float* __restrict__ xg_t,     // [64, 4096]
            const bf16*  __restrict__ hprev,    // [64, 1024] bf16
            const float* __restrict__ done_t,   // [64]
            const bf16*  __restrict__ w_hh,     // [4096, 1024] bf16
            float* __restrict__ c,              // [64, 1024] in/out
            float* __restrict__ hf,             // [64, 1024] f32 out
            bf16*  __restrict__ hnext,          // [64, 1024] bf16 out
            bf16*  __restrict__ hidden_rows)    // [64, 1024] bf16 out (hs slice)
{
    const int H = 1024;
    __shared__ __align__(16) bf16 sA[2][64 * 32];    // masked h tile
    __shared__ __align__(16) bf16 sW[2][128 * 32];   // 4 gates x 32 cols

    const int tid  = threadIdx.x;
    const int lane = tid & 31;
    const int wave = tid >> 5;
    const int j0 = blockIdx.x * 32;
    const int wm = (wave & 3) * 16;
    const int jt = (wave >> 2) * 16;

    // seed accumulators from xg_t (C-layout addressing)
    v8f acc[4];
    {
        const int mb = wm + (lane >> 4) * 8;
        const int nn = j0 + jt + (lane & 15);
#pragma unroll
        for (int q = 0; q < 4; ++q)
#pragma unroll
            for (int r = 0; r < 8; ++r)
                acc[q][r] = xg_t[(size_t)(mb + r) * 4096 + q * H + nn];
    }

    const int arow = tid >> 2;
    const int acol = (tid & 3) * 8;
    const float dmask = done_t[arow];

    auto stageA = [&](int buf, int k0) {     // sync: mask must apply pre-WMMA
        v8bf hv = *(const v8bf*)&hprev[arow * H + k0 + acol];
        if (dmask > 0.5f) {
#pragma unroll
            for (int i = 0; i < 8; ++i) hv[i] = (bf16)0.0f;
        }
        *(v8bf*)&sA[buf][arow * 32 + acol] = hv;
    };
    auto stageW = [&](int buf, int k0) {
#if HAVE_TDM
        if (wave == 0) {
#pragma unroll
            for (int q = 0; q < 4; ++q)
                tdm_load_2d(lds_off(&sW[buf][q * 32 * 32]),
                            w_hh + (size_t)(q * H + j0) * H + k0,
                            (unsigned)(H - k0), (unsigned)(4 * H - (q * H + j0)),
                            (unsigned)H, 32, 32);
        }
#else
#pragma unroll
        for (int rep = 0; rep < 2; ++rep) {
            const int e  = tid + rep * 256;  // 0..511
            const int l  = e >> 2;           // 0..127
            const int cb = (e & 3) * 8;
            const int grow = (l >> 5) * H + j0 + (l & 31);
            stage16(w_hh + (size_t)grow * H + k0 + cb, &sW[buf][l * 32 + cb]);
        }
#endif
    };
    auto xfer_wait = [&]() {
#if HAVE_TDM
        if (wave == 0) tdm_wait();
#else
        stage_wait();
#endif
    };

    stageA(0, 0);
    stageW(0, 0);
    xfer_wait();
    __syncthreads();

    for (int i = 0; i < 32; ++i) {
        const int cur = i & 1;
        if (i + 1 < 32) { stageA(cur ^ 1, (i + 1) * 32); stageW(cur ^ 1, (i + 1) * 32); }

        const int m  = wm + (lane & 15);
        const int ko = (lane >> 4) * 8;
        FragBF a;
        a.h.lo = *(const v8bf*)&sA[cur][m * 32 + ko];
        a.h.hi = *(const v8bf*)&sA[cur][m * 32 + ko + 16];
        const int nb = jt + (lane & 15);
        const int kh = (lane >> 4) * 8;
#pragma unroll
        for (int q = 0; q < 4; ++q) {
            FragBF b;
            b.h.lo = *(const v8bf*)&sW[cur][(q * 32 + nb) * 32 + kh];
            b.h.hi = *(const v8bf*)&sW[cur][(q * 32 + nb) * 32 + kh + 16];
            acc[q] = __builtin_amdgcn_wmma_f32_16x16x32_bf16(
                false, a.v, false, b.v, (short)0, acc[q], false, false);
        }

        if (i + 1 < 32) xfer_wait();
        __syncthreads();
    }

    // epilogue: gate nonlinearities + state update
    const int mb = wm + (lane >> 4) * 8;
    const int nn = j0 + jt + (lane & 15);
#pragma unroll
    for (int r = 0; r < 8; ++r) {
        const int m = mb + r;
        const size_t idx = (size_t)m * H + nn;
        float cold = c[idx];
        if (done_t[m] > 0.5f) cold = 0.0f;
        const float gi = sigmoidf_(acc[0][r]);
        const float gf = sigmoidf_(acc[1][r]);
        const float gg = tanhf(acc[2][r]);
        const float go = sigmoidf_(acc[3][r]);
        const float cn = gf * cold + gi * gg;
        const float hn = go * tanhf(cn);
        c[idx]  = cn;
        hf[idx] = hn;
        hnext[idx] = (bf16)hn;
        hidden_rows[idx] = (bf16)hn;
    }
}

// ---------------------------------------------------------------------------
// logits' = logits + c_w*c_out; log_softmax over A=128; probs/logp_a/entropy
// ---------------------------------------------------------------------------
__global__ void __launch_bounds__(128)
head_combine_k(const float* __restrict__ logits, const float* __restrict__ cwv,
               const float* __restrict__ covv, const int* __restrict__ action,
               float* __restrict__ probs, float* __restrict__ logp_a,
               float* __restrict__ entropy)
{
    __shared__ float red[128];
    const int row = blockIdx.x;
    const int tid = threadIdx.x;
    const size_t idx = (size_t)row * 128 + tid;
    const float l = logits[idx] + cwv[idx] * covv[idx];

    red[tid] = l; __syncthreads();
    for (int s = 64; s > 0; s >>= 1) { if (tid < s) red[tid] = fmaxf(red[tid], red[tid + s]); __syncthreads(); }
    const float mx = red[0]; __syncthreads();

    const float e = expf(l - mx);
    red[tid] = e; __syncthreads();
    for (int s = 64; s > 0; s >>= 1) { if (tid < s) red[tid] += red[tid + s]; __syncthreads(); }
    const float lse = logf(red[0]); __syncthreads();

    const float lp = l - mx - lse;
    const float p  = expf(lp);
    probs[idx] = p;

    red[tid] = p * lp; __syncthreads();
    for (int s = 64; s > 0; s >>= 1) { if (tid < s) red[tid] += red[tid + s]; __syncthreads(); }
    if (tid == 0) entropy[row] = -red[0];
    if (tid == action[row]) logp_a[row] = lp;
}

// ---------------------------------------------------------------------------
// host side
// ---------------------------------------------------------------------------
extern "C" void kernel_launch(void* const* d_in, const int* in_sizes, int n_in,
                              void* d_out, int out_size, void* d_ws, size_t ws_size,
                              hipStream_t stream) {
    (void)in_sizes; (void)n_in; (void)out_size; (void)ws_size;
    const int T = 256, B = 64, OBS = 512, H = 1024, A = 128, TB = T * B;

    const float* x     = (const float*)d_in[0];
    const float* done  = (const float*)d_in[1];
    const float* covh  = (const float*)d_in[2];
    const float* h0    = (const float*)d_in[3];
    const float* c0    = (const float*)d_in[4];
    const int*   actio = (const int*)d_in[5];
    const float* sn_w1 = (const float*)d_in[6];  const float* sn_b1 = (const float*)d_in[7];
    const float* sn_w2 = (const float*)d_in[8];  const float* sn_b2 = (const float*)d_in[9];
    const float* w_ih  = (const float*)d_in[10]; const float* w_hh  = (const float*)d_in[11];
    const float* b_ih  = (const float*)d_in[12]; const float* b_hh  = (const float*)d_in[13];
    const float* cr_w1 = (const float*)d_in[14]; const float* cr_b1 = (const float*)d_in[15];
    const float* cr_w2 = (const float*)d_in[16]; const float* cr_b2 = (const float*)d_in[17];
    const float* ac_w1 = (const float*)d_in[18]; const float* ac_b1 = (const float*)d_in[19];
    const float* ac_w2 = (const float*)d_in[20]; const float* ac_b2 = (const float*)d_in[21];
    const float* cov_w1= (const float*)d_in[22]; const float* cov_b1= (const float*)d_in[23];
    const float* cov_w2= (const float*)d_in[24]; const float* cov_b2= (const float*)d_in[25];
    const float* cw_w1 = (const float*)d_in[26]; const float* cw_b1 = (const float*)d_in[27];
    const float* cw_w2 = (const float*)d_in[28]; const float* cw_b2 = (const float*)d_in[29];

    float* fout = (float*)d_out;
    const size_t OFF_PROBS = 0;
    const size_t OFF_LOGPA = (size_t)TB * A;
    const size_t OFF_ENT   = OFF_LOGPA + TB;
    const size_t OFF_VAL   = OFF_ENT + TB;
    const size_t OFF_HT    = OFF_VAL + TB;
    const size_t OFF_CT    = OFF_HT + (size_t)B * H;

    char* ws = (char*)d_ws;
    size_t off = 0;
    auto alloc = [&](size_t bytes) -> void* {
        void* p = ws + off;
        off = (off + bytes + 255) & ~(size_t)255;
        return p;
    };
    auto cvt = [&](const float* src, size_t n) -> bf16* {
        bf16* dst = (bf16*)alloc(n * sizeof(bf16));
        cvt_f32_bf16_k<<<dim3((unsigned)((n + 255) / 256)), dim3(256), 0, stream>>>(src, dst, (int)n);
        return dst;
    };
    auto gemm = [&](const bf16* Ap, int lda, const bf16* Wp, int ldw, const float* bias,
                    float* outF, bf16* outB, int M, int N, int K, int actv) {
        dim3 g((unsigned)((N + BN - 1) / BN), (unsigned)(M / BM));
        gemm_bf16_wmma_k<<<g, dim3(256), 0, stream>>>(Ap, lda, Wp, ldw, bias, outF, outB,
                                                      M, N, K, actv);
    };

    // ---- bf16 weights & activations ----
    bf16* Wsn1 = cvt(sn_w1, (size_t)H * OBS);
    bf16* Wsn2 = cvt(sn_w2, (size_t)H * H);
    bf16* Wih  = cvt(w_ih,  (size_t)4 * H * H);
    bf16* Whh  = cvt(w_hh,  (size_t)4 * H * H);
    bf16* Wac1 = cvt(ac_w1, (size_t)H * H);
    bf16* Wac2 = cvt(ac_w2, (size_t)A * H);
    bf16* Wcw1 = cvt(cw_w1, (size_t)H * H);
    bf16* Wcw2 = cvt(cw_w2, (size_t)A * H);
    bf16* Wcov1= cvt(cov_w1,(size_t)H * A);
    bf16* Wcov2= cvt(cov_w2,(size_t)A * H);
    bf16* Wcr1 = cvt(cr_w1, (size_t)H * H);
    bf16* Wcr2 = cvt(cr_w2, (size_t)1 * H);

    bf16* Xb   = cvt(x,    (size_t)TB * OBS);
    bf16* Cvb  = cvt(covh, (size_t)TB * A);

    bf16*  tmpB   = (bf16*)alloc((size_t)TB * H * sizeof(bf16));  // t1 / head temps
    bf16*  hidB   = (bf16*)alloc((size_t)TB * H * sizeof(bf16));  // perception output
    bf16*  hiddenB= (bf16*)alloc((size_t)TB * H * sizeof(bf16));  // LSTM hidden states
    float* xg     = (float*)alloc((size_t)TB * 4 * H * sizeof(float));
    float* bsum   = (float*)alloc((size_t)4 * H * sizeof(float));
    float* cbuf   = (float*)alloc((size_t)B * H * sizeof(float));
    float* hfbuf  = (float*)alloc((size_t)B * H * sizeof(float));
    bf16*  hPing  = (bf16*)alloc((size_t)B * H * sizeof(bf16));
    bf16*  hPong  = (bf16*)alloc((size_t)B * H * sizeof(bf16));
    float* logit  = (float*)alloc((size_t)TB * A * sizeof(float));
    float* cwv    = (float*)alloc((size_t)TB * A * sizeof(float));
    float* covv   = (float*)alloc((size_t)TB * A * sizeof(float));

    // ---- perception MLP + hoisted input-gate GEMM ----
    vec_add_k<<<dim3((4 * H + 255) / 256), dim3(256), 0, stream>>>(b_ih, b_hh, bsum, 4 * H);
    gemm(Xb,   OBS, Wsn1, OBS, sn_b1, nullptr, tmpB, TB, H,     OBS, 1);
    gemm(tmpB, H,   Wsn2, H,   sn_b2, nullptr, hidB, TB, H,     H,   1);
    gemm(hidB, H,   Wih,  H,   bsum,  xg,      nullptr, TB, 4*H, H,  0);

    // ---- LSTM scan: 256 sequential fused-step launches ----
    cvt_f32_bf16_k<<<dim3((B * H + 255) / 256), dim3(256), 0, stream>>>(h0, hPing, B * H);
    (void)hipMemcpyAsync(cbuf, c0, (size_t)B * H * sizeof(float), hipMemcpyDeviceToDevice, stream);
    bf16* hcur = hPing; bf16* hnxt = hPong;
    for (int t = 0; t < T; ++t) {
        lstm_step_k<<<dim3(H / 32), dim3(256), 0, stream>>>(
            xg + (size_t)t * B * 4 * H, hcur, done + (size_t)t * B, Whh,
            cbuf, hfbuf, hnxt, hiddenB + (size_t)t * B * H);
        bf16* sw = hcur; hcur = hnxt; hnxt = sw;
    }

    // ---- heads ----
    gemm(hiddenB, H, Wac1, H, ac_b1, nullptr, tmpB, TB, H, H, 1);
    gemm(tmpB,    H, Wac2, H, ac_b2, logit, nullptr, TB, A, H, 0);
    gemm(hiddenB, H, Wcw1, H, cw_b1, nullptr, tmpB, TB, H, H, 1);
    gemm(tmpB,    H, Wcw2, H, cw_b2, cwv,   nullptr, TB, A, H, 0);
    gemm(Cvb,     A, Wcov1,A, cov_b1,nullptr, tmpB, TB, H, A, 1);
    gemm(tmpB,    H, Wcov2,H, cov_b2,covv,  nullptr, TB, A, H, 0);
    gemm(hiddenB, H, Wcr1, H, cr_b1, nullptr, tmpB, TB, H, H, 1);
    gemm(tmpB,    H, Wcr2, H, cr_b2, fout + OFF_VAL, nullptr, TB, 1, H, 0);

    // ---- softmax / entropy / gather + final states ----
    head_combine_k<<<dim3(TB), dim3(128), 0, stream>>>(logit, cwv, covv, actio,
                                                       fout + OFF_PROBS, fout + OFF_LOGPA,
                                                       fout + OFF_ENT);
    (void)hipMemcpyAsync(fout + OFF_HT, hfbuf, (size_t)B * H * sizeof(float),
                         hipMemcpyDeviceToDevice, stream);
    (void)hipMemcpyAsync(fout + OFF_CT, cbuf, (size_t)B * H * sizeof(float),
                         hipMemcpyDeviceToDevice, stream);
}
